// CSDeltaNetLayer_62302795596475
// MI455X (gfx1250) — matrix operand order repaired and checked by reading, not compiled, not run on hardware
//
#include <hip/hip_runtime.h>
#include <hip/hip_bf16.h>
#include <math.h>
#include <stdint.h>

// ---------------- problem constants (from reference) ----------------
#define BATCH  8
#define SEQ    2048
#define DMODEL 512
#define NHEAD  8
#define HDIM   64
#define TOK    (BATCH*SEQ)            // 16384 tokens
#define PCOLS  (NHEAD*HDIM*HDIM)      // 32768 columns of Wperm

typedef __bf16 bf16;
typedef __bf16 v16bf __attribute__((ext_vector_type(16)));
typedef float  v8f   __attribute__((ext_vector_type(8)));

// ---------------- CDNA5 async global->LDS copy (ASYNCcnt tracked) ----------
// the vdst VGPR holds the LDS byte address; generic __shared__ pointers carry the LDS
// byte offset in their low 32 bits, so truncation gives the VDST operand.
__device__ __forceinline__ void async_copy_b128(unsigned lds_byte_off, const void* gptr) {
    asm volatile("global_load_async_to_lds_b128 %0, %1, off"
                 :: "v"(lds_byte_off), "v"((unsigned long long)(uintptr_t)gptr)
                 : "memory");
}
__device__ __forceinline__ void async_wait0() {
    asm volatile("s_wait_asynccnt 0x0" ::: "memory");
}

// ---------------- fragment loaders (CDNA5 WMMA bf16 16x16x32 layouts) ----
// A (16x32, MxK): lane l<16 -> M=l, K={k0..k0+7, k0+16..k0+23}
//                 lane l>=16 -> M=l-16, K={k0+8..k0+15, k0+24..k0+31}
__device__ __forceinline__ v16bf load_a_frag(const bf16* As, int lane, int k0) {
    const int m  = lane & 15;
    const int kb = k0 + ((lane >> 4) << 3);          // +0 or +8
    union { v16bf v; uint4 u[2]; } r;
    r.u[0] = *(const uint4*)(As + m * DMODEL + kb);
    r.u[1] = *(const uint4*)(As + m * DMODEL + kb + 16);
    return r.v;
}
// B (32x16, KxN) from transposed weights Wt[n][k] (K contiguous):
// lanes 0-15: N=l, K={k0..k0+7, k0+16..k0+23} ; lanes 16-31: N=l-16, K={k0+8..k0+15, k0+24..k0+31}
__device__ __forceinline__ v16bf load_b_frag(const bf16* __restrict__ Wt, int lane,
                                             int ncol, int k0) {
    const int kb = k0 + ((lane >> 4) << 3);          // +0 or +8
    const bf16* p = Wt + (size_t)ncol * DMODEL + kb;
    union { v16bf v; uint4 u[2]; } r;
    r.u[0] = *(const uint4*)(p);
    r.u[1] = *(const uint4*)(p + 16);
    return r.v;
}

// ---------------- conversion kernels ----------------
__global__ void f32_to_bf16_kernel(const float* __restrict__ in, bf16* __restrict__ out, int n) {
    for (int i = blockIdx.x * blockDim.x + threadIdx.x; i < n; i += gridDim.x * blockDim.x)
        out[i] = (bf16)in[i];
}
// W is K x N row-major (K=512); Wt is N x 512 (K contiguous per column)
__global__ void transpose_bf16_kernel(const float* __restrict__ W, bf16* __restrict__ Wt, int N) {
    const long long total = (long long)N * DMODEL;
    for (long long e = (long long)blockIdx.x * blockDim.x + threadIdx.x; e < total;
         e += (long long)gridDim.x * blockDim.x) {
        const int n  = (int)(e / DMODEL);
        const int kk = (int)(e % DMODEL);
        Wt[e] = (bf16)W[(size_t)kk * N + n];
    }
}

// ---------------- WMMA GEMM: C[tok][n] = A(bf16 rowmajor, K=512) @ Bt^T ------
// block: 256 thr = 8 waves; grid.x = TOK/16 token tiles; grid.y = N/128.
// EPI: 0 = none, 1 = SiLU.
template <int EPI>
__global__ __launch_bounds__(256) void gemm_k512(const bf16* __restrict__ A,
                                                 const bf16* __restrict__ Bt,
                                                 float* __restrict__ C, int ldc) {
    __shared__ __align__(16) bf16 As[16 * DMODEL];   // 16 KB token tile
    const int tid  = threadIdx.x;
    const int wave = tid >> 5, lane = tid & 31;

    // async-stage 16x512 bf16 tile (16 KB contiguous) into LDS
    {
        const char*    gsrc    = (const char*)(A + (size_t)blockIdx.x * 16 * DMODEL);
        const unsigned ldsbase = (unsigned)(uintptr_t)(void*)As;
#pragma unroll
        for (int i = 0; i < 4; ++i) {
            const unsigned off = (unsigned)(tid + i * 256) * 16u;
            async_copy_b128(ldsbase + off, gsrc + off);
        }
        async_wait0();
    }
    __syncthreads();

    const int ncol = blockIdx.y * 128 + wave * 16 + (lane & 15);
    v8f acc = {};
    for (int k0 = 0; k0 < DMODEL; k0 += 32) {
        if (k0 + 32 < DMODEL)
            __builtin_prefetch(Bt + (size_t)ncol * DMODEL + k0 + 32, 0, 1);
        v16bf a = load_a_frag(As, lane, k0);
        v16bf b = load_b_frag(Bt, lane, ncol, k0);
        acc = __builtin_amdgcn_wmma_f32_16x16x32_bf16(false, a, false, b,
                                                      (short)0, acc, false, false);
    }
    // C/D layout: VGPR r -> row M=r (lanes 0-15) / M=r+8 (lanes 16-31)
    const int mbase = blockIdx.x * 16 + ((lane >> 4) << 3);
#pragma unroll
    for (int r = 0; r < 8; ++r) {
        float v = acc[r];
        if (EPI == 1) v = v / (1.0f + __expf(-v));   // SiLU
        C[(size_t)(mbase + r) * ldc + ncol] = v;
    }
}

// ---------------- Gumbel hash (deterministic stand-in for jax.random.gumbel) --
__device__ __forceinline__ float gumbel_hash(unsigned x) {
    x ^= x >> 17; x *= 0xed5ad4bbu;
    x ^= x >> 11; x *= 0xac4c1b51u;
    x ^= x >> 15; x *= 0x31848babu;
    x ^= x >> 14;
    float u = (float)(x >> 8) * (1.0f / 16777216.0f) + 1e-7f;
    return -__logf(-__logf(u));
}

// ---------------- Wperm GEMM + per-column argmax over rows i --------------
// Forward value of straight-through P is the hard one-hot; softmax is monotone,
// so we only need argmax_i(logits[i][j] + g). Each block: 32 tokens (two 16-row
// M tiles for 2x B-fragment reuse), one head h, one 16-wide j-group, ALL 64
// rows i. Wave w owns i = w*8 .. w*8+7 -> 2x8 accumulator tiles per wave.
__global__ __launch_bounds__(256) void perm_argmax_kernel(const bf16* __restrict__ A,
                                                          const bf16* __restrict__ Wpt,
                                                          unsigned char* __restrict__ idx_out) {
    __shared__ __align__(16) bf16 As[32 * DMODEL];   // 32 KB (two 16-token tiles)
    __shared__ float         lmax[8][32][16];        // 16 KB
    __shared__ unsigned char lidx[8][32][16];        // 4 KB
    const int tid = threadIdx.x, wave = tid >> 5, lane = tid & 31;
    const int h = blockIdx.y, j0 = blockIdx.z * 16;

    // async-stage 32 tokens x 512 bf16 (32 KB contiguous) into LDS
    {
        const char*    gsrc    = (const char*)(A + (size_t)blockIdx.x * 32 * DMODEL);
        const unsigned ldsbase = (unsigned)(uintptr_t)(void*)As;
#pragma unroll
        for (int i = 0; i < 8; ++i) {
            const unsigned off = (unsigned)(tid + i * 256) * 16u;
            async_copy_b128(ldsbase + off, gsrc + off);
        }
        async_wait0();
    }
    __syncthreads();

    v8f acc[2][8];
    const v8f zero = {0.f, 0.f, 0.f, 0.f, 0.f, 0.f, 0.f, 0.f};
#pragma unroll
    for (int u = 0; u < 2; ++u)
#pragma unroll
        for (int t = 0; t < 8; ++t) acc[u][t] = zero;

    const int ibase = wave * 8;
    const int jl    = lane & 15;
    const bf16* As1 = As + 16 * DMODEL;
    for (int k0 = 0; k0 < DMODEL; k0 += 32) {
        v16bf a0 = load_a_frag(As,  lane, k0);
        v16bf a1 = load_a_frag(As1, lane, k0);
#pragma unroll
        for (int t = 0; t < 8; ++t) {
            const int ncol = h * 4096 + (ibase + t) * 64 + j0 + jl;
            v16bf b = load_b_frag(Wpt, lane, ncol, k0);
            acc[0][t] = __builtin_amdgcn_wmma_f32_16x16x32_bf16(false, a0, false, b,
                                                                (short)0, acc[0][t], false, false);
            acc[1][t] = __builtin_amdgcn_wmma_f32_16x16x32_bf16(false, a1, false, b,
                                                                (short)0, acc[1][t], false, false);
        }
    }

    const int mh = (lane >> 4) << 3;
#pragma unroll
    for (int u = 0; u < 2; ++u) {
#pragma unroll
        for (int r = 0; r < 8; ++r) {
            const int m = mh + r;
            const unsigned tok = blockIdx.x * 32u + (unsigned)(u * 16 + m);
            float bestv = -1e30f; int besti = 0;
#pragma unroll
            for (int t = 0; t < 8; ++t) {
                const int i = ibase + t;
                const unsigned key =
                    ((((tok * 8u + (unsigned)h) * 64u + (unsigned)i) * 64u) +
                     (unsigned)(j0 + jl)) ^ 0x9E3779B9u;
                const float v = acc[u][t][r] + gumbel_hash(key);
                if (v > bestv) { bestv = v; besti = i; }
            }
            lmax[wave][u * 16 + m][jl] = bestv;
            lidx[wave][u * 16 + m][jl] = (unsigned char)besti;
        }
    }
    __syncthreads();
    // cross-wave argmax: 512 (m,j) pairs, 256 threads -> 2 each
#pragma unroll
    for (int u = 0; u < 2; ++u) {
        const int mm = (tid >> 4) + u * 16;          // 0..31
        const int jj = tid & 15;
        float bv = lmax[0][mm][jj]; int bi = lidx[0][mm][jj];
#pragma unroll
        for (int w = 1; w < 8; ++w) {
            const float v = lmax[w][mm][jj];
            if (v > bv) { bv = v; bi = lidx[w][mm][jj]; }
        }
        const unsigned tok = blockIdx.x * 32u + (unsigned)mm;
        idx_out[(size_t)tok * DMODEL + h * 64 + j0 + jj] = (unsigned char)bi;
    }
}

// ---------------- per-head L2 norm (in place) ----------------
__global__ void l2norm_kernel(float* __restrict__ t, int nrows) {
    const int row = blockIdx.x * blockDim.x + threadIdx.x;
    if (row >= nrows) return;
    const int tok = row >> 3, hh = row & 7;
    float* p = t + (size_t)tok * DMODEL + hh * 64;
    float s = 0.f;
    for (int i = 0; i < 64; ++i) { const float x = p[i]; s += x * x; }
    const float inv = 1.f / (sqrtf(s) + 1e-6f);
    for (int i = 0; i < 64; ++i) p[i] *= inv;
}

// ---------------- beta = sigmoid(x @ Wbeta) ----------------
__global__ void beta_kernel(const float* __restrict__ x, const float* __restrict__ Wb,
                            float* __restrict__ beta) {
    const int o = blockIdx.x * blockDim.x + threadIdx.x;
    if (o >= TOK * NHEAD) return;
    const int tok = o >> 3, hh = o & 7;
    const float* xr = x + (size_t)tok * DMODEL;
    float s = 0.f;
    for (int kk = 0; kk < DMODEL; ++kk) s += xr[kk] * Wb[kk * NHEAD + hh];
    beta[o] = 1.f / (1.f + __expf(-s));
}

// ---------------- sequential delta-rule scan, one block per (b,h) ----------
// MLD=65: stride 65 == 1 (mod 64 banks) -> column-parallel accesses conflict-free
#define MLD 65
__global__ __launch_bounds__(256) void scan_kernel(const float* __restrict__ q,
                                                   const float* __restrict__ k,
                                                   const float* __restrict__ v,
                                                   const float* __restrict__ beta,
                                                   const unsigned char* __restrict__ idx,
                                                   const int* __restrict__ mask,
                                                   float* __restrict__ out) {
    __shared__ float Mbuf[2][64 * MLD];
    __shared__ float pout[4][64], pdel[4][64];
    __shared__ float qs[64], ks[64], vs[64], ds[64];
    __shared__ float sbeta, smask;
    __shared__ unsigned char pj[64];
    const int tid = threadIdx.x;
    const int b = blockIdx.x >> 3, h = blockIdx.x & 7;

    for (int e = tid; e < 64 * MLD; e += 256) Mbuf[0][e] = 0.f;
    __syncthreads();
    int p = 0;
    for (int t = 0; t < SEQ; ++t) {
        const size_t tok  = (size_t)b * SEQ + t;
        const size_t base = tok * DMODEL + (size_t)h * 64;
        if (tid < 64) {
            qs[tid] = q[base + tid];
            ks[tid] = k[base + tid];
            vs[tid] = v[base + tid];
            pj[tid] = idx[base + tid];
        } else if (tid == 64) {
            sbeta = beta[tok * NHEAD + h];
            smask = (float)mask[tok];
        }
        const int np = p ^ 1;
        for (int e = tid; e < 64 * MLD; e += 256) Mbuf[np][e] = 0.f;
        __syncthreads();
        // M = P @ M : newM[idx[j]][c] += M[j][c]; thread c owns column c
        // (single writer per column -> deterministic, no float atomics)
        if (tid < 64) {
            const int c = tid;
            for (int j = 0; j < 64; ++j)
                Mbuf[np][(int)pj[j] * MLD + c] += Mbuf[p][j * MLD + c];
        }
        __syncthreads();
        p = np;
        // out[d] = sum_h q[h]*M[h][d] ; delta[j] = v[j] - sum_d M[j][d]*k[d]
        // 4-way partial sums across all 256 threads (fixed order -> deterministic)
        {
            const int r0 = tid & 63, qd = tid >> 6;
            float s1 = 0.f, s2 = 0.f;
#pragma unroll
            for (int e2 = 0; e2 < 16; ++e2) {
                const int e = qd * 16 + e2;
                s1 += qs[e] * Mbuf[p][e * MLD + r0];
                s2 += Mbuf[p][r0 * MLD + e] * ks[e];
            }
            pout[qd][r0] = s1;
            pdel[qd][r0] = s2;
        }
        __syncthreads();
        if (tid < 64) {
            out[base + tid] = pout[0][tid] + pout[1][tid] + pout[2][tid] + pout[3][tid];
        } else if (tid < 128) {
            const int j = tid - 64;
            ds[j] = vs[j] - (pdel[0][j] + pdel[1][j] + pdel[2][j] + pdel[3][j]);
        }
        __syncthreads();
        // M[i][j] += beta * k[i] * delta[j] * mask
        const float bm = sbeta * smask;
        for (int e = tid; e < 4096; e += 256) {
            const int i = e >> 6, j = e & 63;
            Mbuf[p][i * MLD + j] += bm * ks[i] * ds[j];
        }
        __syncthreads();
    }
}

// ---------------- RMS norm -> bf16 (ready for Wo GEMM) ----------------
__global__ __launch_bounds__(256) void rms_kernel(const float* __restrict__ in,
                                                  const float* __restrict__ w,
                                                  bf16* __restrict__ outb) {
    __shared__ float red[256];
    __shared__ float sinv;
    const int tok = blockIdx.x, tid = threadIdx.x;
    const float a0 = in[(size_t)tok * DMODEL + tid];
    const float a1 = in[(size_t)tok * DMODEL + 256 + tid];
    red[tid] = a0 * a0 + a1 * a1;
    __syncthreads();
    for (int s = 128; s > 0; s >>= 1) { if (tid < s) red[tid] += red[tid + s]; __syncthreads(); }
    if (tid == 0) sinv = rsqrtf(red[0] * (1.f / (float)DMODEL) + 1e-6f);
    __syncthreads();
    const float inv = sinv;
    outb[(size_t)tok * DMODEL + tid]       = (bf16)(a0 * inv * w[tid]);
    outb[(size_t)tok * DMODEL + 256 + tid] = (bf16)(a1 * inv * w[tid + 256]);
}

// ---------------- host-side orchestration ----------------
extern "C" void kernel_launch(void* const* d_in, const int* in_sizes, int n_in,
                              void* d_out, int out_size, void* d_ws, size_t ws_size,
                              hipStream_t stream) {
    (void)in_sizes; (void)n_in; (void)out_size; (void)ws_size;
    const float* x     = (const float*)d_in[0];
    const float* Wq    = (const float*)d_in[1];
    const float* Wk    = (const float*)d_in[2];
    const float* Wv    = (const float*)d_in[3];
    const float* Wo    = (const float*)d_in[4];
    const float* Wbeta = (const float*)d_in[5];
    const float* Wperm = (const float*)d_in[6];
    const float* rms_w = (const float*)d_in[7];
    const int*   maskp = (const int*)d_in[8];
    float* out = (float*)d_out;

    // workspace carve-up (~212 MB total)
    char* ws = (char*)d_ws;
    size_t off = 0;
    auto alloc = [&](size_t bytes) -> void* {
        void* p = ws + off;
        off += (bytes + 255) & ~(size_t)255;
        return p;
    };
    bf16* xb    = (bf16*)alloc((size_t)TOK * DMODEL * 2);        // 16.8 MB
    bf16* Wqb   = (bf16*)alloc((size_t)DMODEL * DMODEL * 2);
    bf16* Wkb   = (bf16*)alloc((size_t)DMODEL * DMODEL * 2);
    bf16* Wvb   = (bf16*)alloc((size_t)DMODEL * DMODEL * 2);
    bf16* Wob   = (bf16*)alloc((size_t)DMODEL * DMODEL * 2);
    bf16* Wpb   = (bf16*)alloc((size_t)PCOLS * DMODEL * 2);      // 33.6 MB
    float* qbuf = (float*)alloc((size_t)TOK * DMODEL * 4);       // 33.6 MB
    float* kbuf = (float*)alloc((size_t)TOK * DMODEL * 4);
    float* vbuf = (float*)alloc((size_t)TOK * DMODEL * 4);
    float* betb = (float*)alloc((size_t)TOK * NHEAD * 4);
    unsigned char* idxb = (unsigned char*)alloc((size_t)TOK * DMODEL); // 8.4 MB
    float* attn = (float*)alloc((size_t)TOK * DMODEL * 4);
    bf16*  nrmb = (bf16*)alloc((size_t)TOK * DMODEL * 2);

    // 1) precision conversion / weight transposition (one-time, bandwidth-trivial)
    f32_to_bf16_kernel<<<4096, 256, 0, stream>>>(x, xb, TOK * DMODEL);
    transpose_bf16_kernel<<<1024, 256, 0, stream>>>(Wq, Wqb, DMODEL);
    transpose_bf16_kernel<<<1024, 256, 0, stream>>>(Wk, Wkb, DMODEL);
    transpose_bf16_kernel<<<1024, 256, 0, stream>>>(Wv, Wvb, DMODEL);
    transpose_bf16_kernel<<<1024, 256, 0, stream>>>(Wo, Wob, DMODEL);
    transpose_bf16_kernel<<<8192, 256, 0, stream>>>(Wperm, Wpb, PCOLS);

    // 2) projections with fused SiLU (WMMA bf16)
    dim3 gproj(TOK / 16, DMODEL / 128);
    gemm_k512<1><<<gproj, 256, 0, stream>>>(xb, Wqb, qbuf, DMODEL);
    gemm_k512<1><<<gproj, 256, 0, stream>>>(xb, Wkb, kbuf, DMODEL);
    gemm_k512<1><<<gproj, 256, 0, stream>>>(xb, Wvb, vbuf, DMODEL);
    const int nrows = TOK * NHEAD;
    l2norm_kernel<<<(nrows + 255) / 256, 256, 0, stream>>>(qbuf, nrows);
    l2norm_kernel<<<(nrows + 255) / 256, 256, 0, stream>>>(kbuf, nrows);
    beta_kernel<<<(nrows + 255) / 256, 256, 0, stream>>>(x, Wbeta, betb);

    // 3) the 550-GFLOP Wperm GEMM, fused with Gumbel + argmax -> u8 indices
    dim3 gperm(TOK / 32, NHEAD, HDIM / 16);
    perm_argmax_kernel<<<gperm, 256, 0, stream>>>(xb, Wpb, idxb);

    // 4) sequential delta-rule scan (tiny FLOPs, LDS-resident state)
    scan_kernel<<<BATCH * NHEAD, 256, 0, stream>>>(qbuf, kbuf, vbuf, betb, idxb,
                                                   maskp, attn);

    // 5) RMS norm then output projection (WMMA bf16) into d_out
    rms_kernel<<<TOK, 256, 0, stream>>>(attn, rms_w, nrmb);
    gemm_k512<0><<<gproj, 256, 0, stream>>>(nrmb, Wob, out, DMODEL);
}
